// Conv2d_7980049235996
// MI455X (gfx1250) — matrix operand order repaired and checked by reading, not compiled, not run on hardware
//
#include <hip/hip_runtime.h>

typedef __bf16 v2bf  __attribute__((ext_vector_type(2)));
typedef __bf16 v8bf  __attribute__((ext_vector_type(8)));
typedef __bf16 v16bf __attribute__((ext_vector_type(16)));
typedef float  v8f   __attribute__((ext_vector_type(8)));

#define IN_C   128
#define OUT_C  256
#define HW     56
#define HWHW   3136      // 56*56
#define KDIM   1152      // IN_C * 9
#define BM     64
#define BN     128
#define BK     32
#define LDSTR  40        // padded LDS row stride (bf16 elems): 80B -> conflict-free b128

__device__ inline v16bf cat8(v8bf a, v8bf b) {
    return __builtin_shufflevector(a, b, 0, 1, 2, 3, 4, 5, 6, 7,
                                         8, 9, 10, 11, 12, 13, 14, 15);
}

__device__ inline void split2(float v0, float v1, v2bf& h, v2bf& l) {
    const __bf16 h0 = (__bf16)v0, h1 = (__bf16)v1;
    h = (v2bf){h0, h1};
    l = (v2bf){(__bf16)(v0 - (float)h0), (__bf16)(v1 - (float)h1)};
}

__global__ __launch_bounds__(256) void conv_igemm_wmma_bf16x3(
    const float* __restrict__ x,      // [32,128,56,56]
    const float* __restrict__ w,      // [256,128,3,3]
    const float* __restrict__ bias,   // [256]
    float* __restrict__ out)          // [32,256,56,56]
{
    __shared__ __bf16 sAhi[BM * LDSTR];   // weights hi  [m][kk]   kk = ic offset
    __shared__ __bf16 sAlo[BM * LDSTR];
    __shared__ __bf16 sBhi[BN * LDSTR];   // im2col hi   [n][kk]  (K-contiguous)
    __shared__ __bf16 sBlo[BN * LDSTR];

    const int tid    = threadIdx.x;
    const int lane   = tid & 31;
    const int wave   = tid >> 5;      // 0..7
    const int lanelo = lane & 15;
    const int lanehi = lane >> 4;     // 0 or 1
    const int wm     = wave >> 2;     // 0..1 -> M strip (32 rows)
    const int wn     = wave & 3;      // 0..3 -> N strip (32 cols)

    const int n0  = blockIdx.x * BN;  // pixel tile start (100352 % 128 == 0)
    const int oc0 = blockIdx.y * BM;  // out-channel tile start

    // ---- B staging map: thread -> (n, half of ic range) ----
    const int bn_ld  = tid & 127;     // n within tile
    const int bh16   = (tid >> 7) * 16;  // ic sub-offset 0 or 16
    const int p_ld   = n0 + bn_ld;
    const int b_ld   = p_ld / HWHW;
    const int rem_ld = p_ld - b_ld * HWHW;
    const int oh_ld  = rem_ld / HW;
    const int ow_ld  = rem_ld - oh_ld * HW;
    // base of this thread's ic-half, at pixel (oh-1, ow-1):
    const float* xbase = x + b_ld * (IN_C * HWHW) + bh16 * HWHW
                           + (oh_ld - 1) * HW + (ow_ld - 1);

    // ---- A staging map: m = tid>>2 (64 rows), 8 ic per thread ----
    const int am_ld  = tid >> 2;          // 0..63
    const int aq8    = (tid & 3) * 8;     // ic sub-offset 0,8,16,24
    const float* wrow = w + (oc0 + am_ld) * KDIM;

    v8f acc[2][2];
    #pragma unroll
    for (int i = 0; i < 2; ++i)
        #pragma unroll
        for (int j = 0; j < 2; ++j)
            acc[i][j] = (v8f){0.f, 0.f, 0.f, 0.f, 0.f, 0.f, 0.f, 0.f};

    // K' ordering: outer (kh,kw) position, inner 32-channel ic block.
    int kh = 0, kw = 0, icb = 0;
    for (int kt = 0; kt < KDIM / BK; ++kt) {
        const int pos = kh * 3 + kw;      // weight inner offset

        // ---- stage A (global -> regs): 8 stride-9 gathers (L2-resident) ----
        float av[8];
        #pragma unroll
        for (int j = 0; j < 8; ++j)
            av[j] = wrow[(icb + aq8 + j) * 9 + pos];

        // ---- stage B (global -> regs): uniform validity, strided ic reads ----
        const int ih = oh_ld + kh - 1;
        const int iw = ow_ld + kw - 1;
        const bool ok = ((unsigned)ih < HW) && ((unsigned)iw < HW);
        float bv[16];
        #pragma unroll
        for (int j = 0; j < 16; ++j) bv[j] = 0.f;
        if (ok) {
            const float* src = xbase + icb * HWHW + kh * HW + kw;
            #pragma unroll
            for (int j = 0; j < 16; ++j) bv[j] = src[j * HWHW];
        }

        __syncthreads();   // previous compute done reading LDS

        // split + store A (4x v2bf per part)
        #pragma unroll
        for (int jj = 0; jj < 4; ++jj) {
            v2bf h, l;
            split2(av[2 * jj], av[2 * jj + 1], h, l);
            const int off = am_ld * LDSTR + aq8 + 2 * jj;
            *(v2bf*)&sAhi[off] = h;
            *(v2bf*)&sAlo[off] = l;
        }
        // split + store B (8x v2bf per part)
        #pragma unroll
        for (int jj = 0; jj < 8; ++jj) {
            v2bf h, l;
            split2(bv[2 * jj], bv[2 * jj + 1], h, l);
            const int off = bn_ld * LDSTR + bh16 + 2 * jj;
            *(v2bf*)&sBhi[off] = h;
            *(v2bf*)&sBlo[off] = l;
        }

        __syncthreads();   // tiles visible

        // ---- compute: one K=32 step, 12 bf16 WMMAs per wave ----
        v16bf Bh[2], Bl[2];
        #pragma unroll
        for (int nt = 0; nt < 2; ++nt) {
            const int n = wn * 32 + nt * 16 + lanelo;
            const __bf16* pb = &sBhi[n * LDSTR + 8 * lanehi];
            Bh[nt] = cat8(*(const v8bf*)pb, *(const v8bf*)(pb + 16));
            const __bf16* pl = &sBlo[n * LDSTR + 8 * lanehi];
            Bl[nt] = cat8(*(const v8bf*)pl, *(const v8bf*)(pl + 16));
        }
        #pragma unroll
        for (int mt = 0; mt < 2; ++mt) {
            const int m = wm * 32 + mt * 16 + lanelo;
            const __bf16* pa = &sAhi[m * LDSTR + 8 * lanehi];
            v16bf Ah = cat8(*(const v8bf*)pa, *(const v8bf*)(pa + 16));
            const __bf16* pl = &sAlo[m * LDSTR + 8 * lanehi];
            v16bf Al = cat8(*(const v8bf*)pl, *(const v8bf*)(pl + 16));
            #pragma unroll
            for (int nt = 0; nt < 2; ++nt) {
                acc[mt][nt] = __builtin_amdgcn_wmma_f32_16x16x32_bf16(
                    false, Ah, false, Bh[nt], (short)0, acc[mt][nt], false, false);
                acc[mt][nt] = __builtin_amdgcn_wmma_f32_16x16x32_bf16(
                    false, Ah, false, Bl[nt], (short)0, acc[mt][nt], false, false);
                acc[mt][nt] = __builtin_amdgcn_wmma_f32_16x16x32_bf16(
                    false, Al, false, Bh[nt], (short)0, acc[mt][nt], false, false);
            }
        }

        // advance (icb, kw, kh) without div/mod
        icb += BK;
        if (icb == IN_C) {
            icb = 0;
            if (++kw == 3) { kw = 0; ++kh; }
        }
    }

    // ---- epilogue: D layout (reg r -> m = r + 8*lanehi, n = lanelo) ----
    #pragma unroll
    for (int mt = 0; mt < 2; ++mt) {
        #pragma unroll
        for (int nt = 0; nt < 2; ++nt) {
            const int p  = n0 + wn * 32 + nt * 16 + lanelo;
            const int b  = p / HWHW;
            const int pi = p - b * HWHW;
            float* op = out + b * (OUT_C * HWHW) + pi;
            #pragma unroll
            for (int r = 0; r < 8; ++r) {
                const int oc = oc0 + wm * 32 + mt * 16 + r + 8 * lanehi;
                op[oc * HWHW] = acc[mt][nt][r] + bias[oc];
            }
        }
    }
}

extern "C" void kernel_launch(void* const* d_in, const int* in_sizes, int n_in,
                              void* d_out, int out_size, void* d_ws, size_t ws_size,
                              hipStream_t stream) {
    const float* x    = (const float*)d_in[0];
    const float* w    = (const float*)d_in[1];
    const float* bias = (const float*)d_in[2];
    float*       out  = (float*)d_out;

    const int n_pix = 32 * HWHW;              // 100352
    dim3 grid(n_pix / BN, OUT_C / BM);        // 784 x 4
    conv_igemm_wmma_bf16x3<<<grid, 256, 0, stream>>>(x, w, bias, out);
}
